// FMoE_11871289606683
// MI455X (gfx1250) — matrix operand order repaired and checked by reading, not compiled, not run on hardware
//
#include <hip/hip_runtime.h>
#include <hip/hip_bf16.h>
#include <math.h>

// ---------------------------------------------------------------------------
// MoE (top-2, 8 experts) for MI455X / gfx1250.
//   gate -> count/prefix/scatter (compact active pairs) ->
//   FFN1 (x@w1+b1, gelu) -> FFN2 (h@w2+b2) -> weighted combine.
// GEMMs: v_wmma_f32_16x16x32_bf16, double-buffered LDS, B tiles staged by the
// Tensor Data Mover (tensor_load_to_lds + s_wait_tensorcnt), A tiles staged
// manually (token-gathered rows). f32 accumulation. ~232 MB workspace.
// ---------------------------------------------------------------------------

typedef __attribute__((ext_vector_type(16))) __bf16 v16bf;
typedef __attribute__((ext_vector_type(8)))  __bf16 bf16x8;
typedef __attribute__((ext_vector_type(4)))  __bf16 bf16x4;
typedef __attribute__((ext_vector_type(8)))  float  v8f;
typedef unsigned int uint32x4 __attribute__((ext_vector_type(4)));
typedef int          int32x4  __attribute__((ext_vector_type(4)));
typedef int          int32x8  __attribute__((ext_vector_type(8)));

#define NEXPERT 8
#define DMODEL  1024
#define DHIDDEN 4096
#define NTOK    4096
#define TOPK    2
#define NPAIR   (NTOK * TOPK)

// --------------------------- TDM availability ------------------------------
#if defined(__HIP_DEVICE_COMPILE__) && defined(__gfx1250__) && \
    __has_builtin(__builtin_amdgcn_tensor_load_to_lds)
#define USE_TDM 1
#if __has_include(<hip/amd_detail/amd_gfx1250_TDM.h>)
#define TDM_ARGS6 1
#else
#define TDM_ARGS6 0
#endif
#else
#define USE_TDM 0
#endif

// -------------------------------- helpers ----------------------------------

__device__ __forceinline__ v16bf ld_frag(const __bf16* lo, const __bf16* hi) {
  bf16x8 l = *(const bf16x8*)lo;
  bf16x8 h = *(const bf16x8*)hi;
  return __builtin_shufflevector(l, h, 0, 1, 2, 3, 4, 5, 6, 7,
                                       8, 9, 10, 11, 12, 13, 14, 15);
}

__device__ __forceinline__ float gelu_tanh(float v) {
  return 0.5f * v * (1.0f + tanhf(0.7978845608028654f * (v + 0.044715f * v * v * v)));
}

// Issue one TDM 2D tile load: tileRows x 128 bf16 elements from a row-major
// tensor (row pitch = tensorD0 elements) into LDS at ldsOff, with LDS padding
// of 4 DWORDs after every 64 DWORDs (row pitch 256B + 16B = 272B).
__device__ __forceinline__ void tdm_load_tile(unsigned long long gaddr,
                                              unsigned ldsOff, int tensorD0,
                                              int tensorD1, int tileRows) {
#if USE_TDM
  uint32x4 g0;
  g0[0] = 1u;                                   // count=1, user descriptor
  g0[1] = ldsOff;                               // lds_addr
  g0[2] = (unsigned)gaddr;                      // global_addr[31:0]
  g0[3] = (unsigned)((gaddr >> 32) & 0x1FFFFFFull) | (2u << 30);  // [56:32]|type=2
  int32x8 g1;
  g1[0] = (1 << 16) | (1 << 20) | (5 << 22) | (3 << 25);  // 2B data, pad 4dw/64dw
  g1[1] = (tensorD0 & 0xFFFF) << 16;                       // tensor_dim0[15:0]
  g1[2] = ((unsigned)tensorD0 >> 16) | ((tensorD1 & 0xFFFF) << 16);
  g1[3] = ((unsigned)tensorD1 >> 16) | (128 << 16);        // tile_dim0 = 128
  g1[4] = tileRows & 0xFFFF;                               // tile_dim1, tile_dim2=0
  g1[5] = tensorD0;                                        // dim0_stride[31:0]
  g1[6] = 0;
  g1[7] = 0;
  int32x4 z4 = {0, 0, 0, 0};
#if TDM_ARGS6
  int32x8 z8 = {0, 0, 0, 0, 0, 0, 0, 0};
  __builtin_amdgcn_tensor_load_to_lds(g0, g1, z4, z4, z8, 0);
#else
  __builtin_amdgcn_tensor_load_to_lds(g0, g1, z4, z4, 0);
#endif
#endif
}

__device__ __forceinline__ void tdm_wait() {
#if USE_TDM
  __builtin_amdgcn_s_wait_tensorcnt(0);
#endif
}

// ------------------------------ f32 -> bf16 --------------------------------

__global__ void moe_cvt_bf16(const float* __restrict__ in,
                             __bf16* __restrict__ out, long n4) {
  long i = (long)blockIdx.x * blockDim.x + threadIdx.x;
  long stride = (long)gridDim.x * blockDim.x;
  for (; i < n4; i += stride) {
    float4 v = ((const float4*)in)[i];
    bf16x4 o;
    o[0] = (__bf16)v.x; o[1] = (__bf16)v.y; o[2] = (__bf16)v.z; o[3] = (__bf16)v.w;
    ((bf16x4*)out)[i] = o;
  }
}

// --------------------------------- gate ------------------------------------

__global__ void moe_gate(const float* __restrict__ x, const float* __restrict__ gw,
                         const float* __restrict__ gb, int* __restrict__ topIdx,
                         float* __restrict__ topW) {
  const int wave = threadIdx.x >> 5, lane = threadIdx.x & 31;
  const int t = blockIdx.x * 4 + wave;
  const float* xr = x + (size_t)t * DMODEL;
  float acc[NEXPERT] = {};
  for (int d = lane; d < DMODEL; d += 32) {
    float xv = xr[d];
    const float4* g = (const float4*)(gw + (size_t)d * NEXPERT);
    float4 g0 = g[0], g1 = g[1];
    acc[0] += xv * g0.x; acc[1] += xv * g0.y; acc[2] += xv * g0.z; acc[3] += xv * g0.w;
    acc[4] += xv * g1.x; acc[5] += xv * g1.y; acc[6] += xv * g1.z; acc[7] += xv * g1.w;
  }
#pragma unroll
  for (int off = 16; off; off >>= 1)
#pragma unroll
    for (int e = 0; e < NEXPERT; ++e) acc[e] += __shfl_xor(acc[e], off, 32);
  if (lane == 0) {
    float v[NEXPERT];
#pragma unroll
    for (int e = 0; e < NEXPERT; ++e) v[e] = acc[e] + gb[e];
    int i0 = 0; float m0 = v[0];
#pragma unroll
    for (int e = 1; e < NEXPERT; ++e) if (v[e] > m0) { m0 = v[e]; i0 = e; }
    int i1 = (i0 == 0) ? 1 : 0; float m1 = v[i1];
#pragma unroll
    for (int e = 0; e < NEXPERT; ++e)
      if (e != i0 && v[e] > m1) { m1 = v[e]; i1 = e; }
    float e1 = expf(m1 - m0);
    float s = 1.0f + e1;
    topIdx[t * 2 + 0] = i0; topIdx[t * 2 + 1] = i1;
    topW[t * 2 + 0] = 1.0f / s; topW[t * 2 + 1] = e1 / s;
  }
}

// ------------------------- count / prefix / scatter ------------------------

__global__ void moe_count(const int* __restrict__ topIdx, int* __restrict__ cnt) {
  int t = blockIdx.x * blockDim.x + threadIdx.x;
  if (t < NTOK) {
    atomicAdd(&cnt[topIdx[2 * t + 0]], 1);
    atomicAdd(&cnt[topIdx[2 * t + 1]], 1);
  }
}

__global__ void moe_prefix(const int* __restrict__ cnt, int* __restrict__ basep,
                           int* __restrict__ fill) {
  if (threadIdx.x == 0 && blockIdx.x == 0) {
    int s = 0;
    for (int e = 0; e < NEXPERT; ++e) { basep[e] = s; fill[e] = s; s += cnt[e]; }
  }
}

__global__ void moe_scatter(const int* __restrict__ topIdx, int* __restrict__ fill,
                            int* __restrict__ pairTok, int* __restrict__ pairPos) {
  int t = blockIdx.x * blockDim.x + threadIdx.x;
  if (t < NTOK) {
#pragma unroll
    for (int k = 0; k < TOPK; ++k) {
      int e = topIdx[2 * t + k];
      int pos = atomicAdd(&fill[e], 1);
      pairTok[pos] = t;
      pairPos[2 * t + k] = pos;
    }
  }
}

// ------------------------------ FFN GEMM core -------------------------------
// out[p, :] = act(A_row(p) @ W[e] + bias[e])
//   KD = reduction dim, ND = output dim (= W row pitch)
//   GATHER: A rows indexed through pairTok; GELU: fuse gelu + bf16 store.
// 128x128x32 tiles, 8 waves (2x4), wave = 64x32 = 4x2 WMMA 16x16 tiles.
// Double-buffered LDS; B staged by TDM (wave 0), A staged by all threads.

template <int KD, int ND, bool GATHER, bool GELU>
__global__ __launch_bounds__(256)
void moe_ffn(const __bf16* __restrict__ A, const __bf16* __restrict__ W,
             const float* __restrict__ bias, const int* __restrict__ pairTok,
             const int* __restrict__ cnt, const int* __restrict__ basep,
             __bf16* __restrict__ outBF, float* __restrict__ outF) {
  constexpr int BK = 32;
  __shared__ __bf16 As[2][128][40];   // 80B row pitch (16B aligned, staggered)
  __shared__ __bf16 Bs[2][32][136];   // 272B row pitch == TDM pad output

  const int e = blockIdx.z;
  const int cntE = cnt[e], baseE = basep[e];
  const int mBase = blockIdx.y * 128;
  if (mBase >= cntE) return;
  const int nBase = blockIdx.x * 128;
  const int tid = threadIdx.x, lane = tid & 31, wv = tid >> 5;
  const int waveM = wv >> 2, waveN = wv & 3;
  const int half = lane >> 4, mrow = lane & 15;

  const __bf16* wBase = W + (size_t)e * KD * ND + nBase;

  // A staging coords (all 256 threads; 16 bf16 each)
  const int aRow = tid >> 1, aSeg = tid & 1;
  int aIdx = mBase + aRow; if (aIdx >= cntE) aIdx = cntE - 1;
  const __bf16* aSrc;
  if constexpr (GATHER)
    aSrc = A + (size_t)pairTok[baseE + aIdx] * KD + aSeg * 16;
  else
    aSrc = A + (size_t)(baseE + aIdx) * KD + aSeg * 16;

#if !USE_TDM
  const int bRow = tid >> 3, bSeg = tid & 7;   // manual-B fallback coords
#endif

  v8f acc[4][2] = {};

  // ---- stage helpers (inlined via lambdas) ----
  auto stageA = [&](int buf, int k0) {
    const float4* ga = (const float4*)(aSrc + k0);
    float4 a0 = ga[0], a1 = ga[1];
    *reinterpret_cast<float4*>(&As[buf][aRow][aSeg * 16 + 0]) = a0;
    *reinterpret_cast<float4*>(&As[buf][aRow][aSeg * 16 + 8]) = a1;
  };
  auto stageB = [&](int buf, int k0) {
#if USE_TDM
    if (wv == 0) {
      unsigned long long gaddr =
          (unsigned long long)(uintptr_t)(wBase + (size_t)k0 * ND);
      unsigned ldsOff = (unsigned)(uintptr_t)(&Bs[buf][0][0]);
      tdm_load_tile(gaddr, ldsOff, ND, KD, BK);
    }
#else
    const float4* gbv = (const float4*)(wBase + (size_t)(k0 + bRow) * ND + bSeg * 16);
    float4 b0 = gbv[0], b1v = gbv[1];
    *reinterpret_cast<float4*>(&Bs[buf][bRow][bSeg * 16 + 0]) = b0;
    *reinterpret_cast<float4*>(&Bs[buf][bRow][bSeg * 16 + 8]) = b1v;
#endif
  };

  // ---- prologue: fill buffer 0 ----
  stageA(0, 0);
  stageB(0, 0);
  tdm_wait();
  __syncthreads();

  int cur = 0;
  for (int k0 = 0; k0 < KD; k0 += BK) {
    const int nxt = cur ^ 1;

    // 1. fragment loads from current buffer (issue DS loads early)
    v16bf bfrag[2], afrag[4];
#pragma unroll
    for (int nt = 0; nt < 2; ++nt) {
#pragma unroll
      for (int s = 0; s < 16; ++s)
        bfrag[nt][s] = Bs[cur][half * 8 + (s & 7) + (s >> 3) * 16][waveN * 32 + nt * 16 + mrow];
    }
#pragma unroll
    for (int mt = 0; mt < 4; ++mt) {
      const __bf16* ap = &As[cur][waveM * 64 + mt * 16 + mrow][half * 8];
      afrag[mt] = ld_frag(ap, ap + 16);
    }

    // 2. stage next buffer (overlaps with WMMA below)
    if (k0 + BK < KD) {
      stageA(nxt, k0 + BK);
      stageB(nxt, k0 + BK);
      if (k0 + 2 * BK < KD)
        __builtin_prefetch(aSrc + k0 + 2 * BK, 0, 1);
    }

    // 3. matrix ops
#pragma unroll
    for (int mt = 0; mt < 4; ++mt)
#pragma unroll
      for (int nt = 0; nt < 2; ++nt)
        acc[mt][nt] = __builtin_amdgcn_wmma_f32_16x16x32_bf16(
            false, afrag[mt], false, bfrag[nt], (short)0, acc[mt][nt], false, false);

    // 4. pipeline sync: TDM done + all waves done with buffers
    tdm_wait();
    __syncthreads();
    cur = nxt;
  }

  // ---- epilogue: bias (+ gelu) ----
#pragma unroll
  for (int nt = 0; nt < 2; ++nt) {
    const int col = nBase + waveN * 32 + nt * 16 + (lane & 15);
    const float bv = bias[e * ND + col];
#pragma unroll
    for (int mt = 0; mt < 4; ++mt) {
#pragma unroll
      for (int r = 0; r < 8; ++r) {
        int idx = mBase + waveM * 64 + mt * 16 + half * 8 + r;
        if (idx < cntE) {
          float v = acc[mt][nt][r] + bv;
          if constexpr (GELU)
            outBF[(size_t)(baseE + idx) * ND + col] = (__bf16)gelu_tanh(v);
          else
            outF[(size_t)(baseE + idx) * ND + col] = v;
        }
      }
    }
  }
}

// -------------------------------- combine ----------------------------------

__global__ void moe_combine(const float* __restrict__ y, const int* __restrict__ pairPos,
                            const float* __restrict__ topW, float* __restrict__ out) {
  int i = blockIdx.x * blockDim.x + threadIdx.x;  // over NTOK * DMODEL / 4
  int t = i >> 8;                                 // DMODEL/4 = 256 per token
  int d = (i & 255) * 4;
  int p0 = pairPos[t * 2 + 0], p1 = pairPos[t * 2 + 1];
  float w0 = topW[t * 2 + 0], w1 = topW[t * 2 + 1];
  float4 y0 = *(const float4*)(y + (size_t)p0 * DMODEL + d);
  float4 y1 = *(const float4*)(y + (size_t)p1 * DMODEL + d);
  float4 o;
  o.x = w0 * y0.x + w1 * y1.x;
  o.y = w0 * y0.y + w1 * y1.y;
  o.z = w0 * y0.z + w1 * y1.z;
  o.w = w0 * y0.w + w1 * y1.w;
  *(float4*)(out + (size_t)t * DMODEL + d) = o;
}

// ------------------------------- launcher ----------------------------------

extern "C" void kernel_launch(void* const* d_in, const int* in_sizes, int n_in,
                              void* d_out, int out_size, void* d_ws, size_t ws_size,
                              hipStream_t stream) {
  const float* x  = (const float*)d_in[0];
  const float* gw = (const float*)d_in[1];
  const float* gb = (const float*)d_in[2];
  const float* w1 = (const float*)d_in[3];
  const float* b1 = (const float*)d_in[4];
  const float* w2 = (const float*)d_in[5];
  const float* b2 = (const float*)d_in[6];
  float* out = (float*)d_out;

  // workspace carve-up (~232 MB), 256B aligned slices
  char* p = (char*)d_ws;
  auto take = [&](size_t bytes) {
    char* r = p;
    p += (bytes + 255) & ~(size_t)255;
    return r;
  };
  __bf16* xb      = (__bf16*)take((size_t)NTOK * DMODEL * 2);              //   8 MB
  __bf16* w1b     = (__bf16*)take((size_t)NEXPERT * DMODEL * DHIDDEN * 2); //  64 MB
  __bf16* w2b     = (__bf16*)take((size_t)NEXPERT * DHIDDEN * DMODEL * 2); //  64 MB
  __bf16* hbuf    = (__bf16*)take((size_t)NPAIR * DHIDDEN * 2);            //  64 MB
  float*  ybuf    = (float*) take((size_t)NPAIR * DMODEL * 4);             //  32 MB
  int*    topIdx  = (int*)   take((size_t)NTOK * TOPK * 4);
  float*  topW    = (float*) take((size_t)NTOK * TOPK * 4);
  int*    pairTok = (int*)   take((size_t)NPAIR * 4);
  int*    pairPos = (int*)   take((size_t)NTOK * TOPK * 4);
  int*    cntArr  = (int*)   take((size_t)3 * NEXPERT * 4);
  int*    basep   = cntArr + NEXPERT;
  int*    fill    = cntArr + 2 * NEXPERT;

  hipMemsetAsync(cntArr, 0, 3 * NEXPERT * sizeof(int), stream);

  // bf16 conversion passes (bandwidth-bound, ~384 MB total traffic)
  moe_cvt_bf16<<<2048, 256, 0, stream>>>(x, xb, (long)NTOK * DMODEL / 4);
  moe_cvt_bf16<<<4096, 256, 0, stream>>>(w1, w1b, (long)NEXPERT * DMODEL * DHIDDEN / 4);
  moe_cvt_bf16<<<4096, 256, 0, stream>>>(w2, w2b, (long)NEXPERT * DHIDDEN * DMODEL / 4);

  // gating + compaction
  moe_gate<<<NTOK / 4, 128, 0, stream>>>(x, gw, gb, topIdx, topW);
  moe_count<<<NTOK / 256, 256, 0, stream>>>(topIdx, cntArr);
  moe_prefix<<<1, 32, 0, stream>>>(cntArr, basep, fill);
  moe_scatter<<<NTOK / 256, 256, 0, stream>>>(topIdx, fill, pairTok, pairPos);

  // expert FFN GEMMs (grid sized for worst-case expert load; blocks early-exit)
  moe_ffn<DMODEL, DHIDDEN, true, true>
      <<<dim3(DHIDDEN / 128, NTOK / 128, NEXPERT), 256, 0, stream>>>(
          xb, w1b, b1, pairTok, cntArr, basep, hbuf, nullptr);
  moe_ffn<DHIDDEN, DMODEL, false, false>
      <<<dim3(DMODEL / 128, NTOK / 128, NEXPERT), 256, 0, stream>>>(
          hbuf, w2b, b2, nullptr, cntArr, basep, nullptr, ybuf);

  // weighted gather-combine
  moe_combine<<<(NTOK * DMODEL / 4) / 256, 256, 0, stream>>>(ybuf, pairPos, topW, out);
}
